// GatedHippoCell_65171833749528
// MI455X (gfx1250) — matrix operand order repaired and checked by reading, not compiled, not run on hardware
//
#include <hip/hip_runtime.h>
#include <hip/hip_bf16.h>
#include <stdint.h>

// Problem sizes (fixed by the reference)
#define BATCH 128
#define TLEN  512
#define NH    512
#define BM    16            // batch rows per block (one WMMA N tile)
#define NWAVES 16           // waves per block
#define NSL   32            // N columns per wave (2 WMMA tiles)
#define KCHUNK 64           // A_t columns staged per TDM chunk
#define NCHUNK (NH / KCHUNK)

typedef __attribute__((ext_vector_type(16))) __bf16 v16bf;
typedef __attribute__((ext_vector_type(8)))  __bf16 v8bf;
typedef __attribute__((ext_vector_type(8)))  float  v8f;
typedef __attribute__((ext_vector_type(4)))  unsigned int v4u;
typedef __attribute__((ext_vector_type(4)))  int    v4i;
typedef __attribute__((ext_vector_type(8)))  int    v8i;

#if __has_builtin(__builtin_amdgcn_tensor_load_to_lds)
#define HAVE_TDM 1
#else
#define HAVE_TDM 0
#endif

// ---------------------------------------------------------------------------
// TDM: async-copy a 2-D tile A_t[0:512, k0:k0+64] (f32, row stride 512) into
// LDS (contiguous [n][64] layout). D# built per CDNA5 ISA.
// ---------------------------------------------------------------------------
__device__ __forceinline__ void tdm_issue_chunk(const float* gsrc, float* lds_dst) {
#if HAVE_TDM
  unsigned long long ga = (unsigned long long)(uintptr_t)gsrc;
  v4u g0;
  g0.x = 1u;                                   // count=1, user descriptor
  g0.y = (unsigned int)(uintptr_t)lds_dst;     // lds_addr (low 32 bits of generic)
  g0.z = (unsigned int)ga;                     // global_addr[31:0]
  g0.w = (unsigned int)((ga >> 32) & 0x1FFFFFFull) | (2u << 30); // addr[56:32]|type=2
  v8i g1;
  g1[0] = (int)(2u << 16);          // data_size = 4 bytes; no mask/flags/pad
  g1[1] = (int)(KCHUNK << 16);      // tensor_dim0[15:0] = 64
  g1[2] = (int)(NH << 16);          // tensor_dim0 hi=0 | tensor_dim1[15:0]=512
  g1[3] = (int)(KCHUNK << 16);      // tensor_dim1 hi=0 | tile_dim0=64
  g1[4] = (int)NH;                  // tile_dim1=512 | tile_dim2=0
  g1[5] = (int)NH;                  // tensor_dim0_stride lo32 = 512
  g1[6] = 0;                        // stride0 hi | stride1 lo (unused, 2D)
  g1[7] = 0;
  v4i z4 = {0, 0, 0, 0};
#if defined(__clang_major__) && (__clang_major__ >= 23)
  v8i z8 = {0, 0, 0, 0, 0, 0, 0, 0};
  __builtin_amdgcn_tensor_load_to_lds(g0, g1, z4, z4, z8, 0);
#else
  __builtin_amdgcn_tensor_load_to_lds(g0, g1, z4, z4, 0);
#endif
#else
  (void)gsrc; (void)lds_dst;
#endif
}

__device__ __forceinline__ void tdm_wait_le1() {
#if __has_builtin(__builtin_amdgcn_s_wait_tensorcnt)
  __builtin_amdgcn_s_wait_tensorcnt(1);
#else
  asm volatile("s_wait_tensorcnt 0x1" ::: "memory");
#endif
}
__device__ __forceinline__ void tdm_wait_le0() {
#if __has_builtin(__builtin_amdgcn_s_wait_tensorcnt)
  __builtin_amdgcn_s_wait_tensorcnt(0);
#else
  asm volatile("s_wait_tensorcnt 0x0" ::: "memory");
#endif
}

__device__ __forceinline__ v8f wmma_bf16(v16bf a, v16bf b, v8f c) {
  return __builtin_amdgcn_wmma_f32_16x16x32_bf16(false, a, false, b, (short)0, c,
                                                 false, false);
}

// ---------------------------------------------------------------------------
// Prep: Wi/Wh (f32, N x (N+1)) -> bf16 K-major tables in ws, plus fused
// x-column weights w0[n]=Wi[n][0]+Wh[n][0] and bias bsum[n]=bi[n]+bh[n].
// ---------------------------------------------------------------------------
__global__ __launch_bounds__(256) void gh_prep(const float* __restrict__ Wi,
                                               const float* __restrict__ bi,
                                               const float* __restrict__ Wh,
                                               const float* __restrict__ bh,
                                               __bf16* __restrict__ Wib,
                                               __bf16* __restrict__ Whb,
                                               float* __restrict__ w0,
                                               float* __restrict__ bsum) {
  int n = blockIdx.x;
  for (int k = threadIdx.x; k < NH; k += 256) {
    Wib[(size_t)n * NH + k] = (__bf16)Wi[(size_t)n * (NH + 1) + 1 + k];
    Whb[(size_t)n * NH + k] = (__bf16)Wh[(size_t)n * (NH + 1) + 1 + k];
  }
  if (threadIdx.x == 0) {
    w0[n]   = Wi[(size_t)n * (NH + 1)] + Wh[(size_t)n * (NH + 1)];
    bsum[n] = bi[n] + bh[n];
  }
}

// ---------------------------------------------------------------------------
// Scan kernel: 8 persistent blocks x 512 threads. Block owns 16 batch rows;
// c,h live in LDS transposed bf16 [k][b] (= WMMA B-operand layout).
// Per step: gates^T = Wi*c^T + Wh*h^T ; c_new^T = A_t*c^T + f*B_t.
// A_t chunks double-buffered into LDS via TDM.
// ---------------------------------------------------------------------------
__global__ __launch_bounds__(512, 1) void gh_scan(
    const float* __restrict__ x, const float* __restrict__ h0,
    const float* __restrict__ c0, const float* __restrict__ Wf,
    const float* __restrict__ bfp, const float* __restrict__ A_stack,
    const float* __restrict__ B_stack, const __bf16* __restrict__ Wib,
    const __bf16* __restrict__ Whb, const float* __restrict__ w0v,
    const float* __restrict__ bsumv, float* __restrict__ out_hs,
    float* __restrict__ out_cf) {
  __shared__ float  a_sh[2][NH * KCHUNK];   // 2 x 128 KB A_t chunk buffers
  __shared__ __bf16 cT[NH * BM];            // 16 KB, [k][b]
  __shared__ __bf16 hT[NH * BM];            // 16 KB, [k][b]
  __shared__ float  f_sh[BM];
  __shared__ float  x_sh[BM];

  const int tid  = threadIdx.x;
  const int wid  = tid >> 5;
  const int lane = tid & 31;
  const int hlf  = lane >> 4;     // which half of the wave
  const int bcol = lane & 15;     // D-matrix column = batch index in tile
  const int bm   = blockIdx.x * BM;
  const int nbase = wid * NSL;

  // Load initial state h0/c0 into transposed LDS.
  for (int i = tid; i < NH * BM; i += 512) {
    int k = i >> 4, b = i & 15;
    cT[i] = (__bf16)c0[(size_t)(bm + b) * NH + k];
    hT[i] = (__bf16)h0[(size_t)(bm + b) * NH + k];
  }

  // Per-lane time-invariant rows: each lane owns 16 output rows (2 tiles x 8).
  float wf_r[16], w0_r[16], bs_r[16];
  int   nrow[16];
#pragma unroll
  for (int tile = 0; tile < 2; ++tile)
#pragma unroll
    for (int r = 0; r < 8; ++r) {
      int i = tile * 8 + r;
      int n = nbase + tile * 16 + hlf * 8 + r;
      nrow[i] = n;
      wf_r[i] = Wf[1 + n];
      w0_r[i] = w0v[n];
      bs_r[i] = bsumv[n];
    }
  const float wf0 = Wf[0];
  const float bf0 = bfp[0];

#if HAVE_TDM
  if (wid == 0) tdm_issue_chunk(A_stack, &a_sh[0][0]);  // prefetch (t=0, kc=0)
#endif

  for (int t = 0; t < TLEN; ++t) {
    __syncthreads();  // S1: prev-step state writes / f_sh reads complete
    if (tid < BM) {
      float xv = x[(size_t)(bm + tid) * TLEN + t];
      x_sh[tid] = xv;
      f_sh[tid] = bf0 + xv * wf0;  // f accumulator seeded with bias + x term
    }

    v8f accg[2], accc[2];
#pragma unroll
    for (int tt = 0; tt < 2; ++tt)
#pragma unroll
      for (int r = 0; r < 8; ++r) { accg[tt][r] = 0.f; accc[tt][r] = 0.f; }

    const float* At = A_stack + (size_t)t * NH * NH;

    for (int kc = 0; kc < NCHUNK; ++kc) {
      const int cur = kc & 1;
      __syncthreads();  // (A): all consumers of buf[cur^1] are done
#if HAVE_TDM
      if (wid == 0) {
        const bool last = (t == TLEN - 1) && (kc == NCHUNK - 1);
        if (!last) {
          const float* nsrc = (kc < NCHUNK - 1)
                                  ? (At + (size_t)(kc + 1) * KCHUNK)
                                  : (At + (size_t)NH * NH);  // (t+1, kc=0)
          tdm_issue_chunk(nsrc, &a_sh[cur ^ 1][0]);
          tdm_wait_le1();   // everything except the just-issued chunk is done
        } else {
          tdm_wait_le0();
        }
      }
#else
      {  // fallback: cooperative staged copy (no TDM builtin)
        const float* src = At + (size_t)kc * KCHUNK;
        for (int i = tid; i < NH * KCHUNK; i += 512) {
          int n = i / KCHUNK, k = i % KCHUNK;
          a_sh[cur][i] = src[(size_t)n * NH + k];
        }
      }
#endif
      __syncthreads();  // (B): buf[cur] is ready for everyone

      const int kbase = kc * KCHUNK;
#pragma unroll
      for (int ks = 0; ks < KCHUNK; ks += 32) {
        // B operands: lane l holds column bcol, K rows kbase+ks+8*hlf+{0..7, 16..23}.
        const int krow = kbase + ks + hlf * 8;
        v16bf bc, bh;
#pragma unroll
        for (int e = 0; e < 8; ++e) {
          bc[e]     = cT[(krow + e) * BM + bcol];
          bc[e + 8] = cT[(krow + 16 + e) * BM + bcol];
          bh[e]     = hT[(krow + e) * BM + bcol];
          bh[e + 8] = hT[(krow + 16 + e) * BM + bcol];
        }
        const int kb = ks + hlf * 8;  // A-operand K group base for this lane
#pragma unroll
        for (int tile = 0; tile < 2; ++tile) {
          const int row = nbase + tile * 16 + bcol;  // A-operand M row
          const __bf16* wip = Wib + (size_t)row * NH + kbase + kb;
          const __bf16* whp = Whb + (size_t)row * NH + kbase + kb;
          v8bf wi0 = *(const v8bf*)wip;
          v8bf wi1 = *(const v8bf*)(wip + 16);
          v8bf wh0 = *(const v8bf*)whp;
          v8bf wh1 = *(const v8bf*)(whp + 16);
          v16bf awi = __builtin_shufflevector(wi0, wi1, 0, 1, 2, 3, 4, 5, 6, 7,
                                              8, 9, 10, 11, 12, 13, 14, 15);
          v16bf awh = __builtin_shufflevector(wh0, wh1, 0, 1, 2, 3, 4, 5, 6, 7,
                                              8, 9, 10, 11, 12, 13, 14, 15);
          v16bf aat;
          const float* ap = &a_sh[cur][row * KCHUNK + kb];
#pragma unroll
          for (int e = 0; e < 8; ++e) {
            aat[e]     = (__bf16)ap[e];
            aat[e + 8] = (__bf16)ap[16 + e];
          }
          accg[tile] = wmma_bf16(awi, bc, accg[tile]);  // Wi * c^T
          accg[tile] = wmma_bf16(awh, bh, accg[tile]);  // + Wh * h^T
          accc[tile] = wmma_bf16(aat, bc, accc[tile]);  // A_t * c^T
        }
      }
    }  // kc

    // ---- elementwise tail: sigmoid/tanh, f reduction, state update ----
    float hn[16];
    float pf = 0.f;
    const float xb = x_sh[bcol];
    float bst[16];
#pragma unroll
    for (int i = 0; i < 16; ++i) bst[i] = B_stack[(size_t)t * NH + nrow[i]];
#pragma unroll
    for (int tile = 0; tile < 2; ++tile)
#pragma unroll
      for (int r = 0; r < 8; ++r) {
        int i = tile * 8 + r;
        float g = accg[tile][r] + bs_r[i] + xb * w0_r[i];
        float o = 1.f / (1.f + __expf(-g));
        float cold = (float)cT[nrow[i] * BM + bcol];
        float h = o * tanhf(cold);
        hn[i] = h;
        pf += h * wf_r[i];
      }
    atomicAdd(&f_sh[bcol], pf);
    __syncthreads();  // (C): all matmul reads + f accumulation complete

    // h_new -> LDS (next step's B operand) and global hs[b, t, n]
    float* hout = out_hs + ((size_t)(bm + bcol) * TLEN + t) * NH;
#pragma unroll
    for (int tile = 0; tile < 2; ++tile) {
      int n0 = nbase + tile * 16 + hlf * 8;
      float4 v0 = make_float4(hn[tile * 8 + 0], hn[tile * 8 + 1],
                              hn[tile * 8 + 2], hn[tile * 8 + 3]);
      float4 v1 = make_float4(hn[tile * 8 + 4], hn[tile * 8 + 5],
                              hn[tile * 8 + 6], hn[tile * 8 + 7]);
      *(float4*)(hout + n0)     = v0;
      *(float4*)(hout + n0 + 4) = v1;
#pragma unroll
      for (int r = 0; r < 8; ++r)
        hT[(n0 + r) * BM + bcol] = (__bf16)hn[tile * 8 + r];
    }

    const float fb = f_sh[bcol];
#pragma unroll
    for (int tile = 0; tile < 2; ++tile)
#pragma unroll
      for (int r = 0; r < 8; ++r) {
        int i = tile * 8 + r;
        float cn = accc[tile][r] + fb * bst[i];
        cT[nrow[i] * BM + bcol] = (__bf16)cn;
        if (t == TLEN - 1) out_cf[(size_t)(bm + bcol) * NH + nrow[i]] = cn;
      }
  }  // t
}

// ---------------------------------------------------------------------------
extern "C" void kernel_launch(void* const* d_in, const int* in_sizes, int n_in,
                              void* d_out, int out_size, void* d_ws,
                              size_t ws_size, hipStream_t stream) {
  (void)in_sizes; (void)n_in; (void)out_size; (void)ws_size;
  const float* x  = (const float*)d_in[0];
  const float* h0 = (const float*)d_in[1];
  const float* c0 = (const float*)d_in[2];
  const float* Wi = (const float*)d_in[3];
  const float* bi = (const float*)d_in[4];
  const float* Wh = (const float*)d_in[5];
  const float* bh = (const float*)d_in[6];
  const float* Wf = (const float*)d_in[7];
  const float* bf = (const float*)d_in[8];
  const float* A_stack = (const float*)d_in[9];
  const float* B_stack = (const float*)d_in[10];
  float* out = (float*)d_out;

  char* ws = (char*)d_ws;
  __bf16* Wib  = (__bf16*)(ws);
  __bf16* Whb  = (__bf16*)(ws + (size_t)NH * NH * 2);
  float*  w0   = (float*)(ws + (size_t)2 * NH * NH * 2);
  float*  bsum = (float*)(ws + (size_t)2 * NH * NH * 2 + NH * 4);

  hipLaunchKernelGGL(gh_prep, dim3(NH), dim3(256), 0, stream, Wi, bi, Wh, bh,
                     Wib, Whb, w0, bsum);

  float* out_hs = out;
  float* out_cf = out + (size_t)BATCH * TLEN * NH;
  hipLaunchKernelGGL(gh_scan, dim3(BATCH / BM), dim3(512), 0, stream, x, h0, c0,
                     Wf, bf, A_stack, B_stack, Wib, Whb, w0, bsum, out_hs,
                     out_cf);
}
